// CirculantAttention_10909216932595
// MI455X (gfx1250) — matrix operand order repaired and checked by reading, not compiled, not run on hardware
//
#include <hip/hip_runtime.h>
#include <hip/hip_bf16.h>

typedef __attribute__((ext_vector_type(16))) _Float16 v16h;
typedef __attribute__((ext_vector_type(8)))  float    v8f;
typedef __attribute__((ext_vector_type(4)))  unsigned int u32x4;
typedef __attribute__((ext_vector_type(4)))  int i32x4;
typedef __attribute__((ext_vector_type(8)))  int i32x8;

#define BB 8
#define CC 512
#define HH 64
#define WW 64
#define WR 33                 // WW/2+1
#define NTOK (HH*WW)          // 4096
#define MTOK (BB*NTOK)        // 32768
#define FROWS (BB*HH*WR)      // 16896 frequency "tokens" per re/im part
#define TWOPI 6.28318530717958647692f

// ---------------------------------------------------------------------------
// TDM: load a 64-row x 32-half f16 tile (row stride = rowStrideElems halfs in
// global) into LDS at lds_off, padding each 64B row with 16B -> 40-half rows.
// D# layout per CDNA5 ISA (group0/group1).
// ---------------------------------------------------------------------------
__device__ __forceinline__ void tdm_load_tile_64x32_f16(const _Float16* gptr,
                                                        unsigned lds_off,
                                                        int rowStrideElems) {
  unsigned long long ga = (unsigned long long)(size_t)gptr;
  u32x4 g0;
  g0[0] = 1u;                                        // count=1 (valid), load, user
  g0[1] = lds_off;                                   // lds_addr (bytes)
  g0[2] = (unsigned)(ga & 0xFFFFFFFFu);              // global_addr[31:0]
  g0[3] = (unsigned)((ga >> 32) & 0x01FFFFFFu)       // global_addr[56:32]
          | (2u << 30);                              // type=2 ("image")
  i32x8 g1;
  g1[0] = (1 << 16)                                  // data_size=1 -> 2 bytes
        | (1 << 20)                                  // pad_enable
        | (3 << 22)                                  // pad_interval: 16 DWORDs (64B row)
        | (3 << 25);                                 // pad_amount: 4 DWORDs (16B)
  g1[1] = (rowStrideElems & 0xFFFF) << 16;           // tensor_dim0[15:0]
  g1[2] = ((rowStrideElems >> 16) & 0xFFFF)          // tensor_dim0[31:16]
        | (64 << 16);                                // tensor_dim1 = 64 rows
  g1[3] = (32 << 16);                                // tensor_dim1 hi=0, tile_dim0=32
  g1[4] = 64;                                        // tile_dim1=64, tile_dim2=0
  g1[5] = rowStrideElems;                            // tensor_dim0_stride[31:0]
  g1[6] = 0;                                         // stride hi / dim1_stride lo
  g1[7] = 0;
  i32x4 z4 = {0, 0, 0, 0};
#if __clang_major__ >= 23
  i32x8 z8 = {0, 0, 0, 0, 0, 0, 0, 0};
  __builtin_amdgcn_tensor_load_to_lds(g0, g1, z4, z4, z8, 0);
#else
  __builtin_amdgcn_tensor_load_to_lds(g0, g1, z4, z4, 0);
#endif
}

// ---------------------------------------------------------------------------
// Pointwise helpers
// ---------------------------------------------------------------------------

// x:(b,c,n) f32  ->  xf:(b,n,c) f32  and xh:(b,n,c) f16
__global__ void k_transpose_in(const float* __restrict__ x,
                               float* __restrict__ xf, _Float16* __restrict__ xh) {
  size_t idx = (size_t)blockIdx.x * blockDim.x + threadIdx.x;
  if (idx >= (size_t)MTOK * CC) return;
  int c = (int)(idx % CC);
  size_t bn = idx / CC;
  int n = (int)(bn % NTOK);
  int b = (int)(bn / NTOK);
  float v = x[((size_t)b * CC + c) * NTOK + n];
  xf[idx] = v;
  xh[idx] = (_Float16)v;
}

__global__ void k_f32_to_f16(const float* __restrict__ s, _Float16* __restrict__ d, size_t n) {
  size_t i = (size_t)blockIdx.x * blockDim.x + threadIdx.x;
  if (i < n) d[i] = (_Float16)s[i];
}

// yh = f16(y * t)
__global__ void k_mul_cvt(const float* __restrict__ y, const float* __restrict__ t,
                          _Float16* __restrict__ yh, size_t n) {
  size_t i = (size_t)blockIdx.x * blockDim.x + threadIdx.x;
  if (i < n) yh[i] = (_Float16)(y[i] * t[i]);
}

// tok:(b,n,c) -> out:(b,c,n)
__global__ void k_transpose_out(const float* __restrict__ tok, float* __restrict__ out) {
  size_t idx = (size_t)blockIdx.x * blockDim.x + threadIdx.x;
  if (idx >= (size_t)MTOK * CC) return;
  int c = (int)(idx % CC);
  size_t bn = idx / CC;
  int n = (int)(bn % NTOK);
  int b = (int)(bn / NTOK);
  out[((size_t)b * CC + c) * NTOK + n] = tok[idx];
}

// ---------------------------------------------------------------------------
// WMMA GEMM:  out[M,N] = A[M,K] (f16, row-major) * Wt[N,K]^T (f16, row-major)
// optional bias[n] and SiLU. Block = 256 threads (8 waves), tile 64x64,
// K-step 32. Tiles staged into LDS by the Tensor Data Mover (wave 0 issues,
// all waves consume after s_wait_tensorcnt + barrier).
// ---------------------------------------------------------------------------
__global__ void __launch_bounds__(256)
k_wmma_gemm(const _Float16* __restrict__ A, const _Float16* __restrict__ Wt,
            const float* __restrict__ bias, float* __restrict__ out,
            int M, int N, int K, int apply_silu) {
  __shared__ _Float16 sA[64][40];   // 40-half row stride = 80B (16B aligned)
  __shared__ _Float16 sB[64][40];

  const int tid  = threadIdx.x;
  const int wid  = tid >> 5;
  const int lane = tid & 31;
  const int m0 = blockIdx.x * 64;
  const int n0 = blockIdx.y * 64;
  const int wm = (wid & 3) * 16;   // wave's M sub-tile
  const int wn = (wid >> 2) * 32;  // wave's N sub-tile (two 16x16 wmma tiles)

  const unsigned ldsA = (unsigned)(size_t)(&sA[0][0]);  // LDS aperture: low 32b
  const unsigned ldsB = (unsigned)(size_t)(&sB[0][0]);

  v8f acc0 = {}; v8f acc1 = {};

  for (int k0 = 0; k0 < K; k0 += 32) {
    if (wid == 0) {                              // one wave drives the TDM
      tdm_load_tile_64x32_f16(&A[(size_t)m0 * K + k0], ldsA, K);
      tdm_load_tile_64x32_f16(&Wt[(size_t)n0 * K + k0], ldsB, K);
      __builtin_amdgcn_s_wait_tensorcnt(0);
    }
    __syncthreads();

    v16h af, bf0, bf1;
    {
      const int row  = wm + (lane & 15);
      const int koff = (lane >> 4) * 16;          // A: lanes 16..31 hold K=16..31
#pragma unroll
      for (int i = 0; i < 16; ++i) {
        int kk = i + koff;                        // {0..15} or {16..31}
        af[i] = sA[row][kk];
      }
      const int coln  = lane & 15;
      const int kbase = (lane >> 4) * 16;         // B: lanes 16..31 hold K=16..31
#pragma unroll
      for (int i = 0; i < 16; ++i) {
        bf0[i] = sB[wn + coln][kbase + i];
        bf1[i] = sB[wn + 16 + coln][kbase + i];
      }
    }
    acc0 = __builtin_amdgcn_wmma_f32_16x16x32_f16(false, af, false, bf0,
                                                  (short)0, acc0, false, false);
    acc1 = __builtin_amdgcn_wmma_f32_16x16x32_f16(false, af, false, bf1,
                                                  (short)0, acc1, false, false);
    __syncthreads();
  }

  const int nlo = lane & 15;
  const int mhi = (lane >> 4) * 8;                // D: lanes 16..31 hold rows v+8
#pragma unroll
  for (int v = 0; v < 8; ++v) {
    int m = m0 + wm + mhi + v;
    {
      int n = n0 + wn + nlo;
      float r = acc0[v];
      if (bias) r += bias[n];
      if (apply_silu) r = r / (1.0f + __expf(-r));     // x*sigmoid(x)
      out[(size_t)m * N + n] = r;
    }
    {
      int n = n0 + wn + 16 + nlo;
      float r = acc1[v];
      if (bias) r += bias[n];
      if (apply_silu) r = r / (1.0f + __expf(-r));
      out[(size_t)m * N + n] = r;
    }
  }
}

// ---------------------------------------------------------------------------
// DFT passes (layout (B, dim1, dim2, C), transforms along one spatial dim)
// ---------------------------------------------------------------------------

// forward real DFT along w: (B,H,W,C) f32 -> (B,H,WR,C) complex
__global__ void k_dft_w_r2c(const float* __restrict__ src,
                            float* __restrict__ dre, float* __restrict__ dimg,
                            float scale) {
  size_t idx = (size_t)blockIdx.x * blockDim.x + threadIdx.x;
  size_t total = (size_t)BB * HH * WR * CC;
  if (idx >= total) return;
  int c = (int)(idx % CC);
  size_t r = idx / CC;
  int kw = (int)(r % WR); r /= WR;
  int h = (int)(r % HH);
  int b = (int)(r / HH);
  const float* line = src + (((size_t)b * HH + h) * WW) * CC + c;
  float re = 0.f, im = 0.f;
  float w0 = -TWOPI * (float)kw / (float)WW;
  for (int w = 0; w < WW; ++w) {
    float si, co;
    __sincosf(w0 * (float)w, &si, &co);
    float v = line[(size_t)w * CC];
    re += v * co;
    im += v * si;
  }
  dre[idx] = re * scale;
  dimg[idx] = im * scale;
}

// complex DFT along dim1 (length 64) of (B,64,WR,C); sign=-1 fwd, +1 inverse (unnormalized)
__global__ void k_dft64_c2c(const float* __restrict__ sre, const float* __restrict__ sim,
                            float* __restrict__ dre, float* __restrict__ dimg,
                            float sign, float scale) {
  size_t idx = (size_t)blockIdx.x * blockDim.x + threadIdx.x;
  size_t total = (size_t)BB * HH * WR * CC;
  if (idx >= total) return;
  int c = (int)(idx % CC);
  size_t r = idx / CC;
  int kw = (int)(r % WR); r /= WR;
  int p = (int)(r % HH);
  int b = (int)(r / HH);
  size_t base = (((size_t)b * HH) * WR + kw) * CC + c;
  size_t stride = (size_t)WR * CC;
  float re = 0.f, im = 0.f;
  float w0 = sign * TWOPI * (float)p / (float)HH;
  for (int q = 0; q < HH; ++q) {
    float si, co;
    __sincosf(w0 * (float)q, &si, &co);
    float xr = sre[base + (size_t)q * stride];
    float xi = sim[base + (size_t)q * stride];
    re += xr * co - xi * si;
    im += xr * si + xi * co;
  }
  dre[idx] = re * scale;
  dimg[idx] = im * scale;
}

// inverse real DFT along w from half spectrum: (B,H,WR,C) complex -> (B,H,W,C) real
__global__ void k_idft_w_c2r(const float* __restrict__ sre, const float* __restrict__ sim,
                             float* __restrict__ dst, float scale) {
  size_t idx = (size_t)blockIdx.x * blockDim.x + threadIdx.x;
  size_t total = (size_t)BB * HH * WW * CC;
  if (idx >= total) return;
  int c = (int)(idx % CC);
  size_t r = idx / CC;
  int w = (int)(r % WW); r /= WW;
  int h = (int)(r % HH);
  int b = (int)(r / HH);
  size_t base = (((size_t)b * HH + h) * WR) * CC + c;
  float acc = 0.f;
  float w0 = TWOPI * (float)w / (float)WW;
  for (int kw = 0; kw < WR; ++kw) {
    float si, co;
    __sincosf(w0 * (float)kw, &si, &co);
    float term = sre[base + (size_t)kw * CC] * co - sim[base + (size_t)kw * CC] * si;
    acc += (kw == 0 || kw == WR - 1) ? term : 2.f * term;
  }
  dst[idx] = acc * scale;
}

// ---------------------------------------------------------------------------
// Frequency-domain pointwise ops on qkv:(2, FROWS, 1536) (re block, im block)
// ---------------------------------------------------------------------------
__global__ void k_conj_mul_qk(const float* __restrict__ qkv,
                              float* __restrict__ are, float* __restrict__ aim) {
  size_t idx = (size_t)blockIdx.x * blockDim.x + threadIdx.x;
  size_t total = (size_t)FROWS * CC;
  if (idx >= total) return;
  int c = (int)(idx % CC);
  size_t r = idx / CC;
  const float* re = qkv + r * (3 * CC);
  const float* im = qkv + (size_t)FROWS * (3 * CC) + r * (3 * CC);
  float qre = re[c],        qim = im[c];
  float kre = re[CC + c],   kim = im[CC + c];
  are[idx] = qre * kre + qim * kim;                 // Re(conj(q)*k)
  aim[idx] = qre * kim - qim * kre;                 // Im(conj(q)*k)
}

__global__ void k_conj_mul_av(const float* __restrict__ Are, const float* __restrict__ Aim,
                              const float* __restrict__ qkv,
                              float* __restrict__ yre, float* __restrict__ yim) {
  size_t idx = (size_t)blockIdx.x * blockDim.x + threadIdx.x;
  size_t total = (size_t)FROWS * CC;
  if (idx >= total) return;
  int c = (int)(idx % CC);
  size_t r = idx / CC;
  const float* re = qkv + r * (3 * CC);
  const float* im = qkv + (size_t)FROWS * (3 * CC) + r * (3 * CC);
  float vre = re[2 * CC + c], vim = im[2 * CC + c];
  float ar = Are[idx], ai = Aim[idx];
  yre[idx] = ar * vre + ai * vim;                   // Re(conj(A)*v)
  yim[idx] = ar * vim - ai * vre;                   // Im(conj(A)*v)
}

// ---------------------------------------------------------------------------
// Softmax over n (4096 positions) per (b,c); data:(b,n,c) in-place
// ---------------------------------------------------------------------------
__global__ void __launch_bounds__(256) k_softmax_n(float* __restrict__ data) {
  const int bc = blockIdx.x;
  const int b = bc / CC, c = bc % CC;
  float vals[16];
  float m = -1e30f;
#pragma unroll
  for (int i = 0; i < 16; ++i) {
    int n = i * 256 + threadIdx.x;
    float v = data[((size_t)b * NTOK + n) * CC + c];
    vals[i] = v;
    m = fmaxf(m, v);
  }
  __shared__ float sred[256];
  sred[threadIdx.x] = m; __syncthreads();
  for (int s = 128; s > 0; s >>= 1) {
    if (threadIdx.x < s) sred[threadIdx.x] = fmaxf(sred[threadIdx.x], sred[threadIdx.x + s]);
    __syncthreads();
  }
  m = sred[0]; __syncthreads();
  float sum = 0.f;
#pragma unroll
  for (int i = 0; i < 16; ++i) { vals[i] = __expf(vals[i] - m); sum += vals[i]; }
  sred[threadIdx.x] = sum; __syncthreads();
  for (int s = 128; s > 0; s >>= 1) {
    if (threadIdx.x < s) sred[threadIdx.x] += sred[threadIdx.x + s];
    __syncthreads();
  }
  float inv = 1.0f / sred[0];
#pragma unroll
  for (int i = 0; i < 16; ++i) {
    int n = i * 256 + threadIdx.x;
    data[((size_t)b * NTOK + n) * CC + c] = vals[i] * inv;
  }
}

// ---------------------------------------------------------------------------
// Host launch
// ---------------------------------------------------------------------------
static inline size_t align256(size_t x) { return (x + 255) & ~(size_t)255; }
static inline unsigned nb(size_t n, unsigned b) { return (unsigned)((n + b - 1) / b); }

extern "C" void kernel_launch(void* const* d_in, const int* in_sizes, int n_in,
                              void* d_out, int out_size, void* d_ws, size_t ws_size,
                              hipStream_t stream) {
  const float* x      = (const float*)d_in[0];   // (8,512,64,64)
  const float* w_qkv  = (const float*)d_in[1];   // (1536,512)
  const float* w_gate = (const float*)d_in[2];   // (512,512)
  const float* b_gate = (const float*)d_in[3];   // (512)
  const float* w_proj = (const float*)d_in[4];   // (512,512)
  const float* b_proj = (const float*)d_in[5];   // (512)
  float* out = (float*)d_out;

  const size_t NE  = (size_t)MTOK * CC;   // token elements
  const size_t FW  = (size_t)FROWS * CC;  // freq elements (per re/im)

  char* p = (char*)d_ws;
  float*    xf    = (float*)p;    p += align256(NE * 4);
  _Float16* xh    = (_Float16*)p; p += align256(NE * 2);
  float*    t     = (float*)p;    p += align256(NE * 4);
  _Float16* wgh   = (_Float16*)p; p += align256((size_t)CC * CC * 2);
  _Float16* wqh   = (_Float16*)p; p += align256((size_t)3 * CC * CC * 2);
  _Float16* wph   = (_Float16*)p; p += align256((size_t)CC * CC * 2);
  float*    F1re  = (float*)p;    p += align256(FW * 4);
  float*    F1im  = (float*)p;    p += align256(FW * 4);
  float*    F2re  = (float*)p;    p += align256(FW * 4);
  float*    F2im  = (float*)p;    p += align256(FW * 4);
  float*    Xf    = (float*)p;    p += align256(2 * FW * 4);   // re block || im block
  _Float16* Xh    = (_Float16*)p; p += align256(2 * FW * 2);
  float*    qkv   = (float*)p;    p += align256((size_t)2 * FROWS * 3 * CC * 4);
  float*    attn  = (float*)p;    p += align256(NE * 4);       // reused for proj out
  float*    y     = (float*)p;    p += align256(NE * 4);
  _Float16* yh    = (_Float16*)p; p += align256(NE * 2);
  (void)ws_size; (void)in_sizes; (void)n_in; (void)out_size;

  // 1. transpose input to token-major + f16 copy
  k_transpose_in<<<nb(NE, 256), 256, 0, stream>>>(x, xf, xh);

  // 2. weight f16 copies
  k_f32_to_f16<<<nb((size_t)CC * CC, 256), 256, 0, stream>>>(w_gate, wgh, (size_t)CC * CC);
  k_f32_to_f16<<<nb((size_t)3 * CC * CC, 256), 256, 0, stream>>>(w_qkv, wqh, (size_t)3 * CC * CC);
  k_f32_to_f16<<<nb((size_t)CC * CC, 256), 256, 0, stream>>>(w_proj, wph, (size_t)CC * CC);

  // 3. gate: t = silu(xf @ Wg^T + bg)    [WMMA + TDM]
  k_wmma_gemm<<<dim3(MTOK / 64, CC / 64), 256, 0, stream>>>(xh, wgh, b_gate, t,
                                                           MTOK, CC, CC, 1);

  // 4-5. X = rfft2(xf, ortho): w-pass then h-pass (scale 1/64)
  k_dft_w_r2c<<<nb(FW, 256), 256, 0, stream>>>(xf, F1re, F1im, 1.0f);
  k_dft64_c2c<<<nb(FW, 256), 256, 0, stream>>>(F1re, F1im, Xf, Xf + FW, -1.0f, 1.0f / 64.0f);

  // 6-7. qkv = X @ Wqkv^T (complex*real == stacked re/im rows)   [WMMA + TDM]
  k_f32_to_f16<<<nb(2 * FW, 256), 256, 0, stream>>>(Xf, Xh, 2 * FW);
  k_wmma_gemm<<<dim3(2 * FROWS / 64, (3 * CC) / 64), 256, 0, stream>>>(
      Xh, wqh, nullptr, qkv, 2 * FROWS, 3 * CC, CC, 0);

  // 8. attn_freq = conj(q)*k
  k_conj_mul_qk<<<nb(FW, 256), 256, 0, stream>>>(qkv, F1re, F1im);

  // 9-10. attn = irfft2(attn_freq, ortho): inverse h (c2c,+1) then c2r (scale 1/64)
  k_dft64_c2c<<<nb(FW, 256), 256, 0, stream>>>(F1re, F1im, F2re, F2im, 1.0f, 1.0f);
  k_idft_w_c2r<<<nb(NE, 256), 256, 0, stream>>>(F2re, F2im, attn, 1.0f / 64.0f);

  // 11. softmax over n per (b,c)
  k_softmax_n<<<BB * CC, 256, 0, stream>>>(attn);

  // 12-13. A = rfft2(attn) (backward norm -> no scaling)
  k_dft_w_r2c<<<nb(FW, 256), 256, 0, stream>>>(attn, F1re, F1im, 1.0f);
  k_dft64_c2c<<<nb(FW, 256), 256, 0, stream>>>(F1re, F1im, F2re, F2im, -1.0f, 1.0f);

  // 14. y_freq = conj(A)*v
  k_conj_mul_av<<<nb(FW, 256), 256, 0, stream>>>(F2re, F2im, qkv, F1re, F1im);

  // 15-16. y = irfft2(y_freq, ortho)
  k_dft64_c2c<<<nb(FW, 256), 256, 0, stream>>>(F1re, F1im, F2re, F2im, 1.0f, 1.0f);
  k_idft_w_c2r<<<nb(NE, 256), 256, 0, stream>>>(F2re, F2im, y, 1.0f / 64.0f);

  // 17. y *= t, convert to f16
  k_mul_cvt<<<nb(NE, 256), 256, 0, stream>>>(y, t, yh, NE);

  // 18. proj: tok = yh @ Wp^T + bp   [WMMA + TDM]  (reuse attn buffer)
  k_wmma_gemm<<<dim3(MTOK / 64, CC / 64), 256, 0, stream>>>(yh, wph, b_proj, attn,
                                                           MTOK, CC, CC, 0);

  // 19. transpose (b,n,c) -> (b,c,h,w)
  k_transpose_out<<<nb(NE, 256), 256, 0, stream>>>(attn, out);
}
